// CrossAttention_56427280335239
// MI455X (gfx1250) — matrix operand order repaired and checked by reading, not compiled, not run on hardware
//
#include <hip/hip_runtime.h>
#include <hip/hip_bf16.h>

// CDNA5 / gfx1250: wave32, WMMA bf16 16x16x32 with f32 accumulation.
typedef __attribute__((ext_vector_type(16))) __bf16 v16bf;
typedef __attribute__((ext_vector_type(8)))  float  v8f;

#define DEV __device__ __forceinline__

// round-to-nearest-even f32 -> bf16 (inputs are finite Gaussians; no NaN path)
DEV unsigned int pack_bf16x2(float lo, float hi) {
  unsigned int a = __float_as_uint(lo);
  unsigned int b = __float_as_uint(hi);
  a = (a + 0x7FFFu + ((a >> 16) & 1u)) >> 16;
  b = (b + 0x7FFFu + ((b >> 16) & 1u)) >> 16;
  return a | (b << 16);
}
DEV unsigned short f32_to_bf16(float f) {
  unsigned int a = __float_as_uint(f);
  return (unsigned short)((a + 0x7FFFu + ((a >> 16) & 1u)) >> 16);
}

union Frag {
  unsigned int u[8];
  uint4 q[2];
  v16bf v;
};

// Problem constants
// B=8 query imgs, N=16 support imgs, D=512, DK=128, U=V=784 = 49 tiles of 16.
#define NTILE 49

// ---------------------------------------------------------------------------
// Stage 1: channel projections out[k,u] = sum_d W[k,d] * X[d,u] per image.
// One wave per (img, k-tile of 16, u-tile of 16); K=512 -> 16 bf16 WMMAs.
// Epilogue scatters results into WMMA-fragment-packed layouts for later GEMMs:
//   MODE 0: qk  -> A-packed with row=u, K=k   (scores GEMM A operand)
//   MODE 1: qv  -> B-packed with N=u,  K=k   (+ b2[u] = sum_k qv^2 via atomics)
//   MODE 2: sk  -> B-packed with N=v,  K=k   (scores GEMM B operand)
//   MODE 3: sv  -> plain f32 [n][k][v]       (weighted-sum stage)
// Packed layout: [img][tile16][kc(4)][lane(32)][16 bf16], 32B/lane contiguous.
// ---------------------------------------------------------------------------
template<int MODE>
__global__ __launch_bounds__(32)
void proj_kernel(const float* __restrict__ X, const float* __restrict__ W,
                 unsigned short* __restrict__ outPk, float* __restrict__ outF,
                 float* __restrict__ b2)
{
  const int lane = threadIdx.x;
  const int ut   = blockIdx.x;   // u tile 0..48
  const int mt   = blockIdx.y;   // k tile 0..7
  const int img  = blockIdx.z;
  const int lo   = lane & 15;
  const int hiL  = lane >> 4;

  const float* Ximg = X + (size_t)img * 512 * 784;
  const int u    = ut * 16 + lo;
  const int wrow = mt * 16 + lo;

  v8f acc = {0.f,0.f,0.f,0.f,0.f,0.f,0.f,0.f};
  #pragma unroll 4
  for (int kc = 0; kc < 16; ++kc) {
    // A fragment: W row (M=lane%16), K pairs per ISA 16-bit A layout
    const float* wp = W + (size_t)wrow * 512 + kc * 32 + hiL * 8;
    float4 w0 = *(const float4*)(wp);
    float4 w1 = *(const float4*)(wp + 4);
    float4 w2 = *(const float4*)(wp + 16);
    float4 w3 = *(const float4*)(wp + 20);
    Frag A;
    A.u[0] = pack_bf16x2(w0.x, w0.y); A.u[1] = pack_bf16x2(w0.z, w0.w);
    A.u[2] = pack_bf16x2(w1.x, w1.y); A.u[3] = pack_bf16x2(w1.z, w1.w);
    A.u[4] = pack_bf16x2(w2.x, w2.y); A.u[5] = pack_bf16x2(w2.z, w2.w);
    A.u[6] = pack_bf16x2(w3.x, w3.y); A.u[7] = pack_bf16x2(w3.z, w3.w);
    // B fragment: X column u, K rows (stride-784 gathers; image is L2-resident)
    const int d0 = kc * 32 + hiL * 8;
    const float* xp = Ximg + (size_t)d0 * 784 + u;
    Frag Bf;
    #pragma unroll
    for (int j = 0; j < 8; ++j)
      Bf.u[j] = pack_bf16x2(xp[(2*j + (j >= 4 ? 8 : 0)) * 784], xp[(2*j + 1 + (j >= 4 ? 8 : 0)) * 784]);
    acc = __builtin_amdgcn_wmma_f32_16x16x32_bf16(false, A.v, false, Bf.v,
                                                  (short)0, acc, false, false);
  }

  // C layout: lane holds N=u fixed, M rows k = mt*16 + r + 8*hiL
  #pragma unroll
  for (int r = 0; r < 8; ++r) {
    const int k = mt * 16 + r + 8 * hiL;
    const float val = acc[r];
    if (MODE == 0) {                       // A-packed, row=u, Kdim=k
      const int kc2 = k >> 5, kk = k & 31;
      const int hi  = (kk >> 3) & 1;
      const int lp  = lo + 16 * hi;
      const int idx = (kk & 7) + 8 * (kk >> 4);
      outPk[((((size_t)img * NTILE + ut) * 4 + kc2) * 32 + lp) * 16 + idx] = f32_to_bf16(val);
    } else if (MODE == 1 || MODE == 2) {   // B-packed, N=u, Kdim=k
      const int kc2 = k >> 5, kk = k & 31;
      const int hi  = (kk >> 3) & 1;
      const int lp  = lo + 16 * hi;
      const int idx = (kk & 7) + 8 * (kk >> 4);
      outPk[((((size_t)img * NTILE + ut) * 4 + kc2) * 32 + lp) * 16 + idx] = f32_to_bf16(val);
    } else {                               // plain f32 [img][k][u]
      outF[((size_t)img * 128 + k) * 784 + u] = val;
    }
  }
  if (MODE == 1) {
    float ssq = 0.f;
    #pragma unroll
    for (int r = 0; r < 8; ++r) ssq += acc[r] * acc[r];
    atomicAdd(b2 + (size_t)img * 784 + u, ssq);   // 16 partials per entry
  }
}

// ---------------------------------------------------------------------------
// Stage 2 (dominant, 20 GFLOP): scores[b,n,u,v] = qk^T sk / sqrt(128),
// softmax over n (16 accumulators live in registers), reduced over u on the
// fly -> attnsum[b,n,v]. One wave per (b, v-tile, u-chunk of 7 tiles).
// 64 v_wmma per u-tile; softmax VALU co-executes with XDL WMMA.
// ---------------------------------------------------------------------------
__global__ __launch_bounds__(32)
void scores_kernel(const unsigned short* __restrict__ qkA,
                   const unsigned short* __restrict__ skB,
                   float* __restrict__ attnsum)
{
  const int lane = threadIdx.x;
  const int uc = blockIdx.x, vt = blockIdx.y, b = blockIdx.z;
  // exp(s/sqrt(128)) == exp2(s * log2e/sqrt(128)); scores ~N(0,1) so no max-sub
  constexpr float kScale = 0.08838834764831845f * 1.4426950408889634f;

  float asum[16];
  #pragma unroll
  for (int n = 0; n < 16; ++n) asum[n] = 0.f;

  for (int ui = 0; ui < 7; ++ui) {
    const int ut = uc * 7 + ui;
    Frag Af[4];
    #pragma unroll
    for (int kc = 0; kc < 4; ++kc) {
      const uint4* p = (const uint4*)(qkA + ((((size_t)b * NTILE + ut) * 4 + kc) * 32 + lane) * 16);
      Af[kc].q[0] = p[0]; Af[kc].q[1] = p[1];
    }
    v8f sc[16];
    #pragma unroll
    for (int n = 0; n < 16; ++n) {
      v8f c = {0.f,0.f,0.f,0.f,0.f,0.f,0.f,0.f};
      #pragma unroll
      for (int kc = 0; kc < 4; ++kc) {
        Frag Bf;
        const uint4* p = (const uint4*)(skB + ((((size_t)n * NTILE + vt) * 4 + kc) * 32 + lane) * 16);
        Bf.q[0] = p[0]; Bf.q[1] = p[1];
        c = __builtin_amdgcn_wmma_f32_16x16x32_bf16(false, Af[kc].v, false, Bf.v,
                                                    (short)0, c, false, false);
      }
      sc[n] = c;
    }
    // per C slot (u = r + 8*hiL within tile, v = lane%16): softmax over n
    #pragma unroll
    for (int r = 0; r < 8; ++r) {
      float e[16], s = 0.f;
      #pragma unroll
      for (int n = 0; n < 16; ++n) { e[n] = __builtin_amdgcn_exp2f(sc[n][r] * kScale); s += e[n]; }
      const float rs = __builtin_amdgcn_rcpf(s);
      #pragma unroll
      for (int n = 0; n < 16; ++n) asum[n] += e[n] * rs;
    }
  }
  const int v = vt * 16 + (lane & 15);
  #pragma unroll
  for (int n = 0; n < 16; ++n)
    atomicAdd(attnsum + ((size_t)b * 16 + n) * 784 + v, asum[n]);
}

// ---------------------------------------------------------------------------
// Stage 3 (25 MFLOP): qa[b,k,v] = sum_n attnsum[b,n,v] * sv[n,k,v].
// Writes qa in A-packed bf16 (row=v) for the distance GEMM, and
// a2[b,v] = sum_k qa^2 via block reduction (unique writer, no atomics).
// ---------------------------------------------------------------------------
__global__ __launch_bounds__(128)
void qa_kernel(const float* __restrict__ attnsum, const float* __restrict__ svF,
               unsigned short* __restrict__ qaA, float* __restrict__ a2)
{
  const int k = threadIdx.x;   // 0..127
  const int v = blockIdx.x;    // 0..783
  const int b = blockIdx.y;    // 0..7
  float qa = 0.f;
  #pragma unroll
  for (int n = 0; n < 16; ++n) {
    const float w = attnsum[((size_t)b * 16 + n) * 784 + v]; // uniform -> scalar loads
    qa += w * svF[((size_t)n * 128 + k) * 784 + v];
  }
  const int kc = k >> 5, kk = k & 31;
  const int hi  = (kk >> 3) & 1;
  const int lp  = (v & 15) + 16 * hi;
  const int idx = (kk & 7) + 8 * (kk >> 4);
  qaA[((((size_t)b * NTILE + (v >> 4)) * 4 + kc) * 32 + lp) * 16 + idx] = f32_to_bf16(qa);

  __shared__ float red[128];
  red[k] = qa * qa;
  __syncthreads();
  #pragma unroll
  for (int s = 64; s > 0; s >>= 1) {
    if (k < s) red[k] += red[k + s];
    __syncthreads();
  }
  if (k == 0) a2[(size_t)b * 784 + v] = red[0];
}

// ---------------------------------------------------------------------------
// Stage 4 (1.3 GFLOP): ab[b,v,u] = sum_k qa[k,v] qv[k,u]; fused clamp-distance
// reduction straight to out[b]. One wave per (b, v-tile, u-chunk of 7).
// ---------------------------------------------------------------------------
__global__ __launch_bounds__(32)
void dist_kernel(const unsigned short* __restrict__ qaA,
                 const unsigned short* __restrict__ qvB,
                 const float* __restrict__ a2,
                 const float* __restrict__ b2,
                 float* __restrict__ out)
{
  const int lane = threadIdx.x;
  const int uc = blockIdx.x, vt = blockIdx.y, b = blockIdx.z;
  const int lo = lane & 15, hiL = lane >> 4;

  Frag Af[4];
  #pragma unroll
  for (int kc = 0; kc < 4; ++kc) {
    const uint4* p = (const uint4*)(qaA + ((((size_t)b * NTILE + vt) * 4 + kc) * 32 + lane) * 16);
    Af[kc].q[0] = p[0]; Af[kc].q[1] = p[1];
  }
  float a2v[8];
  #pragma unroll
  for (int r = 0; r < 8; ++r) a2v[r] = a2[(size_t)b * 784 + vt * 16 + r + 8 * hiL];

  float part = 0.f;
  for (int ui = 0; ui < 7; ++ui) {
    const int ut = uc * 7 + ui;
    v8f c = {0.f,0.f,0.f,0.f,0.f,0.f,0.f,0.f};
    #pragma unroll
    for (int kc = 0; kc < 4; ++kc) {
      Frag Bf;
      const uint4* p = (const uint4*)(qvB + ((((size_t)b * NTILE + ut) * 4 + kc) * 32 + lane) * 16);
      Bf.q[0] = p[0]; Bf.q[1] = p[1];
      c = __builtin_amdgcn_wmma_f32_16x16x32_bf16(false, Af[kc].v, false, Bf.v,
                                                  (short)0, c, false, false);
    }
    const float b2u = b2[(size_t)b * 784 + ut * 16 + lo];
    #pragma unroll
    for (int r = 0; r < 8; ++r)
      part += fmaxf(a2v[r] + b2u - 2.f * c[r], 0.f);
  }
  #pragma unroll
  for (int off = 16; off >= 1; off >>= 1)
    part += __shfl_xor(part, off, 32);
  if (lane == 0) atomicAdd(out + b, part * (1.0f / (784.0f * 784.0f)));
}

// ---------------------------------------------------------------------------
extern "C" void kernel_launch(void* const* d_in, const int* in_sizes, int n_in,
                              void* d_out, int out_size, void* d_ws, size_t ws_size,
                              hipStream_t stream)
{
  (void)in_sizes; (void)n_in; (void)ws_size;
  const float* query   = (const float*)d_in[0];   // 8 x 512 x 784
  const float* support = (const float*)d_in[1];   // 16 x 512 x 784
  const float* Wk      = (const float*)d_in[2];   // 128 x 512
  const float* Wv      = (const float*)d_in[3];   // 128 x 512
  float* out = (float*)d_out;                     // 8

  char* ws = (char*)d_ws;
  size_t off = 0;
  auto take = [&](size_t bytes) -> char* {
    char* p = ws + off;
    off += (bytes + 255) & ~(size_t)255;
    return p;
  };
  unsigned short* qkA = (unsigned short*)take(8ull  * NTILE * 4 * 512 * 2); // 1.6 MB
  unsigned short* qvB = (unsigned short*)take(8ull  * NTILE * 4 * 512 * 2); // 1.6 MB
  unsigned short* skB = (unsigned short*)take(16ull * NTILE * 4 * 512 * 2); // 3.2 MB
  unsigned short* qaA = (unsigned short*)take(8ull  * NTILE * 4 * 512 * 2); // 1.6 MB
  float* svF     = (float*)take(16ull * 128 * 784 * 4);                     // 6.4 MB
  float* attnsum = (float*)take(8ull * 16 * 784 * 4);                       // 0.4 MB
  float* a2      = (float*)take(8ull * 784 * 4);
  float* b2      = (float*)take(8ull * 784 * 4);

  hipMemsetAsync(attnsum, 0, 8ull * 16 * 784 * 4, stream);
  hipMemsetAsync(b2,      0, 8ull * 784 * 4,      stream);
  hipMemsetAsync(out,     0, (size_t)out_size * sizeof(float), stream);

  const dim3 blk32(32, 1, 1);
  proj_kernel<0><<<dim3(NTILE, 8, 8),  blk32, 0, stream>>>(query,   Wk, qkA, nullptr, nullptr);
  proj_kernel<1><<<dim3(NTILE, 8, 8),  blk32, 0, stream>>>(query,   Wv, qvB, nullptr, b2);
  proj_kernel<2><<<dim3(NTILE, 8, 16), blk32, 0, stream>>>(support, Wk, skB, nullptr, nullptr);
  proj_kernel<3><<<dim3(NTILE, 8, 16), blk32, 0, stream>>>(support, Wv, nullptr, svF, nullptr);

  scores_kernel<<<dim3(7, NTILE, 8), blk32, 0, stream>>>(qkA, skB, attnsum);
  qa_kernel<<<dim3(784, 8, 1), dim3(128, 1, 1), 0, stream>>>(attnsum, svF, qaA, a2);
  dist_kernel<<<dim3(7, NTILE, 8), blk32, 0, stream>>>(qaA, qvB, a2, b2, out);
}
